// TextConditionEncoder_72816875536936
// MI455X (gfx1250) — matrix operand order repaired and checked by reading, not compiled, not run on hardware
//
#include <hip/hip_runtime.h>
#include <hip/hip_bf16.h>
#include <cstdint>

typedef __attribute__((ext_vector_type(16))) __bf16       v16bf;
typedef __attribute__((ext_vector_type(8)))  float        v8f;
typedef __attribute__((ext_vector_type(4)))  unsigned int v4u;
typedef __attribute__((ext_vector_type(8)))  int          v8i;
typedef __attribute__((ext_vector_type(4)))  int          v4i;

// ---------------------------------------------------------------------------
// helpers
// ---------------------------------------------------------------------------
__device__ __forceinline__ uint32_t pack_bf16x2(float a, float b) {
    union { float f; uint32_t u; } ca, cb;
    ca.f = a; cb.f = b;
    // round-half-up to bf16, splice high halves with one v_perm_b32:
    // D = (a.u>>16) | (b.u & 0xFFFF0000) -> a in [15:0] (even K), b in [31:16]
    uint32_t ua = ca.u + 0x8000u;
    uint32_t ub = cb.u + 0x8000u;
    return __builtin_amdgcn_perm(ub, ua, 0x07060302u);
}

union Frag { uint32_t u[8]; v16bf v; };

// TDM: DMA a 2-D tile (128 rows x 32 dwords) global -> LDS (ISA ch.8 D#).
// row_stride in dwords; tile rows land contiguously at lds_addr (stride 32 dw).
__device__ __forceinline__ void tdm_load_tile(const uint32_t* gptr,
                                              uint32_t lds_addr_bytes,
                                              int row_stride_dw) {
    uint64_t ga = (uint64_t)(uintptr_t)gptr;
    v4u g0;
    g0[0] = 1u;                                              // count=1 (valid user D#)
    g0[1] = lds_addr_bytes;                                  // lds_addr
    g0[2] = (uint32_t)ga;                                    // global_addr[31:0]
    g0[3] = (uint32_t)((ga >> 32) & 0x1FFFFFFu) | (2u << 30);// addr[56:32] | type=2
    v8i g1;
    g1[0] = (int)(2u << 16);          // data_size=4B; no multicast/pad/iterate
    g1[1] = (int)(32u << 16);         // tensor_dim0[15:0]=32  (atomic_barrier=0)
    g1[2] = (int)(128u << 16);        // tensor_dim0 hi=0, tensor_dim1[15:0]=128
    g1[3] = (int)(32u << 16);         // tensor_dim1 hi=0, tile_dim0=32
    g1[4] = (int)128u;                // tile_dim1=128, tile_dim2=0 (2-D)
    g1[5] = row_stride_dw;            // tensor_dim0_stride[31:0]
    g1[6] = 0;                        // stride hi=0, tensor_dim1_stride lo=0
    g1[7] = 0;
    v4i z4 = {0, 0, 0, 0};
    v8i z8 = {0, 0, 0, 0, 0, 0, 0, 0};
    __builtin_amdgcn_tensor_load_to_lds(g0, g1, z4, z4, z8, 0);
}

// ---------------------------------------------------------------------------
// Fused GEMM:  out[M,N] = (res? res : 0) + A[M,K] @ W[K,N] + bias[N]
//  ABF16=true : A is pre-packed bf16 pairs (uint32[M][K/2]); tile fetched by
//               the Tensor Data Mover straight into LDS (double-buffered,
//               s_wait_tensorcnt), zero VGPR staging for A.
//  ABF16=false: A is fp32 (converted in-kernel; used only for the text GEMM).
// Block: 256 threads (8 wave32), tile 128(M) x 64(N), K-step 64
// (2 unrolled WMMA sub-steps -> 8 v_wmma per wave per barrier group).
// W tile register double-buffer: global b128 loads of tile k+1 in flight
// during the WMMAs of tile k. M%128==0, K%64==0, N%32==0 for all calls.
// ---------------------------------------------------------------------------
template<bool ABF16>
__global__ __launch_bounds__(256)
void gemm_wmma(const void* __restrict__ Av, const float* __restrict__ W,
               const float* __restrict__ bias, const float* __restrict__ res,
               float* __restrict__ out, int M, int N, int K) {
    __shared__ uint32_t lds_a[2][128 * 32]; // packed bf16 pairs [m][kpair] x2  32 KB
    __shared__ float    lds_braw[64 * 68];  // raw fp32 W tile [k][n], pad=4    17 KB
    __shared__ uint32_t lds_b[64 * 32];     // packed bf16 pairs [n][kpair]      8 KB

    const int tid  = threadIdx.x;
    const int lane = tid & 31;
    const int wv   = tid >> 5;
    const int wm   = wv >> 1;          // 0..3  (M sub-tile)
    const int wn   = wv & 1;           // 0..1  (N sub-tile)
    const int lm   = lane & 15;
    const int lg   = lane >> 4;        // lane group 0/1

    const int m0 = blockIdx.y * 128;
    const int n0 = blockIdx.x * 64;

    const uint32_t* Ab = (const uint32_t*)Av;   // ABF16 view
    const float*    Af = (const float*)Av;      // fp32 view

    v8f acc[2][2] = {};

    float4 areg[8];    // staged A quads  (fp32 path only)
    float4 breg[4];    // staged W quads

    auto issue_b = [&](int k0) {
        #pragma unroll
        for (int j = 0; j < 4; ++j) {
            int q   = tid + j * 256;          // quad index 0..1023
            int kr  = q >> 4;                 // 0..63
            int nc4 = q & 15;
            int gn  = n0 + nc4 * 4;
            breg[j] = (gn < N)                // N % 32 == 0 -> quad fully in/out
                    ? *(const float4*)&W[(size_t)(k0 + kr) * N + gn]
                    : make_float4(0.f, 0.f, 0.f, 0.f);
        }
    };
    auto issue_a_f32 = [&](int k0) {
        #pragma unroll
        for (int j = 0; j < 8; ++j) {
            int q   = tid + j * 256;          // quad index 0..2047
            int row = q >> 4;                 // 0..127
            int qc  = q & 15;
            areg[j] = *(const float4*)&Af[(size_t)(m0 + row) * K + k0 + qc * 4];
        }
    };

    // ---- prologue: first tiles in flight ----
    if constexpr (ABF16) {
        if (wv == 0)
            tdm_load_tile(&Ab[(size_t)m0 * (K >> 1)],
                          (uint32_t)(uintptr_t)&lds_a[0][0], K >> 1);
    } else {
        issue_a_f32(0);
    }
    issue_b(0);

    for (int k0 = 0; k0 < K; k0 += 64) {
        const int pk = (k0 >> 6) & 1;
        __syncthreads();   // previous iteration's fragment reads complete

        // ---- commit staged A tile to LDS (fp32 path only) ----
        if constexpr (!ABF16) {
            #pragma unroll
            for (int j = 0; j < 8; ++j) {
                int q   = tid + j * 256;
                int row = q >> 4;
                int qc  = q & 15;
                uint2 p;
                p.x = pack_bf16x2(areg[j].x, areg[j].y);
                p.y = pack_bf16x2(areg[j].z, areg[j].w);
                *(uint2*)&lds_a[0][row * 32 + qc * 2] = p;
            }
        }
        // ---- commit staged W tile raw fp32 to LDS ----
        #pragma unroll
        for (int j = 0; j < 4; ++j) {
            int q   = tid + j * 256;
            int kr  = q >> 4;
            int nc4 = q & 15;
            *(float4*)&lds_braw[kr * 68 + nc4 * 4] = breg[j];
        }
        if constexpr (ABF16) {
            if (wv == 0) __builtin_amdgcn_s_wait_tensorcnt(0);  // A tile pk landed
        }
        __syncthreads();   // braw + A tile visible to all waves

        // ---- repack W tile: [k][n] fp32 -> [n][kpair] bf16 (once per block) ----
        #pragma unroll
        for (int j = 0; j < 8; ++j) {
            int q  = tid + j * 256;           // pair index 0..2047
            int n  = q & 63;                  // n fastest -> conflict-free braw reads
            int kp = q >> 6;                  // k-pair 0..31
            float b0 = lds_braw[(kp * 2 + 0) * 68 + n];
            float b1 = lds_braw[(kp * 2 + 1) * 68 + n];
            lds_b[n * 32 + kp] = pack_bf16x2(b0, b1);
        }

        // ---- kick off next tile's fetches; they fly during the WMMAs ----
        if (k0 + 64 < K) {
            issue_b(k0 + 64);
            if constexpr (ABF16) {
                if (wv == 0)
                    tdm_load_tile(&Ab[(size_t)m0 * (K >> 1) + ((k0 + 64) >> 1)],
                                  (uint32_t)(uintptr_t)&lds_a[pk ^ 1][0], K >> 1);
            } else {
                issue_a_f32(k0 + 64);
            }
        }

        __syncthreads();   // lds_b ready

        // ---- two WMMA sub-steps (K = 32 each) from the staged tiles ----
        const uint32_t* abase = ABF16 ? &lds_a[pk][0] : &lds_a[0][0];
        #pragma unroll
        for (int s = 0; s < 2; ++s) {
            Frag af[2], bf[2];
            #pragma unroll
            for (int r = 0; r < 2; ++r) {
                const uint32_t* ap = abase + (wm * 32 + r * 16 + lm) * 32 + s * 16;
                int kb2 = lg * 4;
                #pragma unroll
                for (int i = 0; i < 4; ++i) af[r].u[i] = ap[kb2 + i];      // K = lg*8 + 2i
                #pragma unroll
                for (int i = 4; i < 8; ++i) af[r].u[i] = ap[kb2 + i + 4];  // K = lg*8 + 16 + 2(i-4)
            }
            #pragma unroll
            for (int c = 0; c < 2; ++c) {
                const uint32_t* bp = &lds_b[(wn * 32 + c * 16 + lm) * 32 + s * 16 + lg * 4];
                #pragma unroll
                for (int j = 0; j < 8; ++j) bf[c].u[j] = bp[j + (j & 4)];  // K = lg*8 + 2j (+8 for j>=4)
            }
            #pragma unroll
            for (int r = 0; r < 2; ++r)
                #pragma unroll
                for (int c = 0; c < 2; ++c)
                    acc[r][c] = __builtin_amdgcn_wmma_f32_16x16x32_bf16(
                        false, af[r].v, false, bf[c].v, (short)0, acc[r][c], false, false);
        }
    }

    // ---- epilogue: + bias (+ residual); C layout: VGPR rr -> M = rr + 8*lg ----
    #pragma unroll
    for (int r = 0; r < 2; ++r) {
        #pragma unroll
        for (int c = 0; c < 2; ++c) {
            int n = n0 + wn * 32 + c * 16 + lm;
            if (n < N) {
                float bn = bias[n];
                #pragma unroll
                for (int rr = 0; rr < 8; ++rr) {
                    int m = m0 + wm * 32 + r * 16 + rr + lg * 8;
                    size_t o = (size_t)m * N + n;
                    float v = acc[r][c][rr] + bn;
                    if (res) v += res[o];
                    out[o] = v;
                }
            }
        }
    }
}

// ---------------------------------------------------------------------------
// Activation pre-pack: fp32 -> packed bf16 pairs, optional ReLU.
// One float4 -> uint2 per thread. count4 = total_elems / 4.
// ---------------------------------------------------------------------------
__global__ __launch_bounds__(256)
void pack_kernel(const float* __restrict__ in, uint32_t* __restrict__ out,
                 int count4, int relu) {
    int i = blockIdx.x * 256 + threadIdx.x;
    if (i >= count4) return;
    float4 v = *(const float4*)&in[(size_t)i * 4];
    if (relu) {
        v.x = __builtin_fmaxf(v.x, 0.f); v.y = __builtin_fmaxf(v.y, 0.f);
        v.z = __builtin_fmaxf(v.z, 0.f); v.w = __builtin_fmaxf(v.w, 0.f);
    }
    uint2 p; p.x = pack_bf16x2(v.x, v.y); p.y = pack_bf16x2(v.z, v.w);
    *(uint2*)&out[(size_t)i * 2] = p;
}

// ---------------------------------------------------------------------------
// VQ: one 64-thread block per (b, hw) position. argmin over 1024 codes with
// first-index tie-break, write zq, accumulate sum((zq - z)^2) into lacc.
// ---------------------------------------------------------------------------
__global__ __launch_bounds__(64)
void vq_kernel(const float* __restrict__ z, const float* __restrict__ cb,
               float* __restrict__ zq, float* __restrict__ lacc) {
    __shared__ float zsh[64];
    __shared__ float dsh[64];
    __shared__ int   ish[64];
    const int row = blockIdx.x;
    const int t   = threadIdx.x;
    zsh[t] = z[(size_t)row * 64 + t];
    __syncthreads();

    float best = 3.4e38f; int bi = 0;
    for (int j = 0; j < 16; ++j) {
        int c = t + j * 64;
        const float* e = cb + (size_t)c * 64;
        float d = 0.f;
        #pragma unroll
        for (int k = 0; k < 64; k += 4) {
            const float4 ev = *(const float4*)&e[k];
            float d0 = zsh[k + 0] - ev.x;
            float d1 = zsh[k + 1] - ev.y;
            float d2 = zsh[k + 2] - ev.z;
            float d3 = zsh[k + 3] - ev.w;
            d = fmaf(d0, d0, d); d = fmaf(d1, d1, d);
            d = fmaf(d2, d2, d); d = fmaf(d3, d3, d);
        }
        if (d < best) { best = d; bi = c; }   // c strictly increases -> keeps first min
    }
    dsh[t] = best; ish[t] = bi;
    __syncthreads();
    for (int s = 32; s > 0; s >>= 1) {
        if (t < s) {
            float d2 = dsh[t + s]; int i2 = ish[t + s];
            if (d2 < dsh[t] || (d2 == dsh[t] && i2 < ish[t])) { dsh[t] = d2; ish[t] = i2; }
        }
        __syncthreads();
    }
    const int idx = ish[0];
    float q  = cb[(size_t)idx * 64 + t];
    zq[(size_t)row * 64 + t] = q;             // straight-through forward value
    float df = q - zsh[t];
    __syncthreads();
    dsh[t] = df * df;
    __syncthreads();
    for (int s = 32; s > 0; s >>= 1) {
        if (t < s) dsh[t] += dsh[t + s];
        __syncthreads();
    }
    if (t == 0) atomicAdd(lacc, dsh[0]);
}

__global__ void zero_kernel(float* p) { p[0] = 0.f; }

__global__ void loss_kernel(const float* __restrict__ lacc, float* __restrict__ out) {
    // vq_loss = mean(sq) + BETA*mean(sq) = 1.25 * sum / (B*HW*C)
    out[0] = 1.25f * lacc[0] / (float)(256 * 64 * 64);
}

// ---------------------------------------------------------------------------
// launch helpers
// ---------------------------------------------------------------------------
static inline void launch_gemm_f32A(const float* A, const float* W, const float* b,
                                    const float* res, float* out,
                                    int M, int N, int K, hipStream_t s) {
    dim3 grid((N + 63) / 64, (M + 127) / 128);
    gemm_wmma<false><<<grid, 256, 0, s>>>((const void*)A, W, b, res, out, M, N, K);
}
static inline void launch_gemm_bf16A(const uint32_t* A, const float* W, const float* b,
                                     const float* res, float* out,
                                     int M, int N, int K, hipStream_t s) {
    dim3 grid((N + 63) / 64, (M + 127) / 128);
    gemm_wmma<true><<<grid, 256, 0, s>>>((const void*)A, W, b, res, out, M, N, K);
}
static inline void launch_pack(const float* in, uint32_t* out, int elems, int relu,
                               hipStream_t s) {
    int count4 = elems / 4;
    pack_kernel<<<(count4 + 255) / 256, 256, 0, s>>>(in, out, count4, relu);
}

extern "C" void kernel_launch(void* const* d_in, const int* in_sizes, int n_in,
                              void* d_out, int out_size, void* d_ws, size_t ws_size,
                              hipStream_t stream) {
    constexpr int SXD = 2048, HWC = 4096, BATCH = 256;
    const float* text   = (const float*)d_in[0];
    const float* w_in   = (const float*)d_in[1];
    const float* b_in   = (const float*)d_in[2];
    const float* a0w    = (const float*)d_in[3];
    const float* a0b    = (const float*)d_in[4];
    const float* a1w    = (const float*)d_in[5];
    const float* a1b    = (const float*)d_in[6];
    const float* w_tovq = (const float*)d_in[7];
    const float* b_tovq = (const float*)d_in[8];
    const float* cb     = (const float*)d_in[9];
    const float* c0w    = (const float*)d_in[10];
    const float* c0b    = (const float*)d_in[11];
    const float* c1w    = (const float*)d_in[12];
    const float* c1b    = (const float*)d_in[13];
    const float* w_out  = (const float*)d_in[14];
    const float* b_out  = (const float*)d_in[15];
    float* out = (float*)d_out;

    float*    ws   = (float*)d_ws;
    float*    x    = ws;                               // 256*2048 fp32
    float*    h    = x  + (size_t)BATCH * SXD;         // 256*4096 fp32 (shared temp)
    float*    y    = h  + (size_t)BATCH * HWC;         // 256*4096 fp32 (z)
    float*    zq   = y  + (size_t)BATCH * HWC;         // 256*4096 fp32
    float*    lacc = zq + (size_t)BATCH * HWC;         // 1 (+3 pad)
    uint32_t* xb   = (uint32_t*)(lacc + 4);            // 256*1024 packed pairs
    uint32_t* hb   = xb + (size_t)BATCH * (SXD / 2);   // 256*2048 packed pairs
    uint32_t* zqb  = hb + (size_t)BATCH * (HWC / 2);   // 256*2048 packed pairs

    zero_kernel<<<1, 1, 0, stream>>>(lacc);

    // x = text @ w_in + b_in            (16384 x 32), flat == (256, 2048)
    launch_gemm_f32A(text, w_in, b_in, nullptr, x, BATCH * 64, 32, 1024, stream);

    // resnet-a: net = relu(x)@w0+b0 ; x += relu(net)@w1+b1
    for (int i = 0; i < 4; ++i) {
        launch_pack(x, xb, BATCH * SXD, 1, stream);
        launch_gemm_bf16A(xb, a0w + (size_t)i * SXD * SXD, a0b + (size_t)i * SXD,
                          nullptr, h, BATCH, SXD, SXD, stream);
        launch_pack(h, hb, BATCH * SXD, 1, stream);
        launch_gemm_bf16A(hb, a1w + (size_t)i * SXD * SXD, a1b + (size_t)i * SXD,
                          x, x, BATCH, SXD, SXD, stream);
    }

    // y = x @ w_tovq + b_tovq           (256 x 4096), no ReLU on x
    launch_pack(x, xb, BATCH * SXD, 0, stream);
    launch_gemm_bf16A(xb, w_tovq, b_tovq, nullptr, y, BATCH, HWC, SXD, stream);

    // VQ nearest-codebook + loss accumulation
    vq_kernel<<<BATCH * 64, 64, 0, stream>>>(y, cb, zq, lacc);

    // resnet-c on zq (in place)
    for (int i = 0; i < 4; ++i) {
        launch_pack(zq, zqb, BATCH * HWC, 1, stream);
        launch_gemm_bf16A(zqb, c0w + (size_t)i * HWC * HWC, c0b + (size_t)i * HWC,
                          nullptr, h, BATCH, HWC, HWC, stream);
        launch_pack(h, hb, BATCH * HWC, 1, stream);
        launch_gemm_bf16A(hb, c1w + (size_t)i * HWC * HWC, c1b + (size_t)i * HWC,
                          zq, zq, BATCH, HWC, HWC, stream);
    }

    // text_hat -> d_out[1..] ; loss -> d_out[0]
    launch_pack(zq, zqb, BATCH * HWC, 0, stream);
    launch_gemm_bf16A(zqb, w_out, b_out, nullptr, out + 1, BATCH, 1024, HWC, stream);
    loss_kernel<<<1, 1, 0, stream>>>(lacc, out);
}
